// SVDHead_22187801051262
// MI455X (gfx1250) — matrix operand order repaired and checked by reading, not compiled, not run on hardware
//
#include <hip/hip_runtime.h>
#include <hip/hip_bf16.h>

#define BATCH 16
#define NPTS  2048
#define DKDIM 512

typedef __bf16 bf16x8  __attribute__((ext_vector_type(8)));
typedef __bf16 bf16x16 __attribute__((ext_vector_type(16)));
typedef float  f32x8   __attribute__((ext_vector_type(8)));
typedef int    v4i     __attribute__((ext_vector_type(4)));

#if defined(__AMDGCN__) && __has_builtin(__builtin_amdgcn_global_load_async_to_lds_b128) && __has_builtin(__builtin_amdgcn_s_wait_asynccnt)
#define USE_ASYNC_LDS 1
#else
#define USE_ASYNC_LDS 0
#endif

static __device__ __forceinline__ bf16x16 combine8(bf16x8 a, bf16x8 b) {
    return __builtin_shufflevector(a, b, 0,1,2,3,4,5,6,7,8,9,10,11,12,13,14,15);
}

static __device__ __forceinline__ unsigned short f2bf(float f) {
    unsigned int u = __float_as_uint(f);
    unsigned int r = (u + 0x7FFFu + ((u >> 16) & 1u)) >> 16;   // round-to-nearest-even
    return (unsigned short)r;
}

static __device__ __forceinline__ float redmax16(float v) {
    #pragma unroll
    for (int o = 8; o >= 1; o >>= 1) v = fmaxf(v, __shfl_xor(v, o, 32));
    return v;
}
static __device__ __forceinline__ float redsum16(float v) {
    #pragma unroll
    for (int o = 8; o >= 1; o >>= 1) v += __shfl_xor(v, o, 32);
    return v;
}

// ---------------------------------------------------------------------------
// Kernel 0: zero the per-batch reduction buffer (16 batches x 16 floats)
// ---------------------------------------------------------------------------
__global__ void zero_red_kernel(float* __restrict__ red) {
    red[threadIdx.x] = 0.0f;
}

// ---------------------------------------------------------------------------
// Kernel 1: transpose + convert f32 [B, DK, N] -> bf16 [B, N, DK]
// grid (N/32, DK/32, 2*B), block (32, 8)
// ---------------------------------------------------------------------------
__global__ __launch_bounds__(256) void transpose_bf16_kernel(
    const float* __restrict__ src_emb, const float* __restrict__ tgt_emb,
    unsigned short* __restrict__ QT, unsigned short* __restrict__ KT)
{
    __shared__ float tile[32][33];
    const int which = blockIdx.z >> 4;       // 0 = src->QT, 1 = tgt->KT
    const int b     = blockIdx.z & 15;
    const float*          in  = which ? tgt_emb : src_emb;
    unsigned short*       out = which ? KT      : QT;
    const int n0 = blockIdx.x * 32, d0 = blockIdx.y * 32;
    const float* ib = in + (size_t)b * DKDIM * NPTS;
    #pragma unroll
    for (int i = 0; i < 4; ++i) {
        int d = d0 + threadIdx.y + i * 8;
        tile[threadIdx.y + i * 8][threadIdx.x] = ib[(size_t)d * NPTS + n0 + threadIdx.x];
    }
    __syncthreads();
    unsigned short* ob = out + (size_t)b * NPTS * DKDIM;
    #pragma unroll
    for (int i = 0; i < 4; ++i) {
        int n = n0 + threadIdx.y + i * 8;
        ob[(size_t)n * DKDIM + d0 + threadIdx.x] = f2bf(tile[threadIdx.x][threadIdx.y + i * 8]);
    }
}

// ---------------------------------------------------------------------------
// Kernel 2: fused flash-attention style scores+softmax+src_corr+reductions.
// grid (N/128, B), block 256 (8 waves). Wave w owns rows [blk*128+w*16, +16).
// A (Q rows, bf16) in registers; K tile (16 cols x DK bf16) double-buffered
// in LDS via CDNA5 async-to-LDS loads (ASYNCcnt) and shared by all 8 waves;
// 16x chained v_wmma_f32_16x16x32_bf16 per 16x16 score tile.
// ---------------------------------------------------------------------------
__global__ __launch_bounds__(256) void svdhead_main_kernel(
    const unsigned short* __restrict__ QT, const unsigned short* __restrict__ KT,
    const float* __restrict__ src, const float* __restrict__ tgt,
    const float* __restrict__ temperature, float* __restrict__ red)
{
    __shared__ alignas(16) unsigned short ktile[2][16 * DKDIM];   // 2 x 16 KB
    __shared__ float tgts[3 * NPTS];                              // 24 KB

    const int b    = blockIdx.y;
    const int tid  = threadIdx.x;
    const int wave = tid >> 5;
    const int lane = tid & 31;
    const int half = lane >> 4;      // 0: lanes 0-15, 1: lanes 16-31
    const int lr   = lane & 15;

    const float scale = temperature[b] * 0.044194173824159216f;  // temp / sqrt(512)

    // stage V (= tgt points) in LDS
    const float* tgtb = tgt + (size_t)b * 3 * NPTS;
    for (int i = tid; i < 3 * NPTS; i += 256) tgts[i] = tgtb[i];

    // preload the wave's 16x512 bf16 Q tile as 16 A-fragments (registers)
    const int row_base = blockIdx.x * 128 + wave * 16;
    const unsigned short* arow = QT + ((size_t)b * NPTS + row_base + lr) * DKDIM;
    bf16x16 afrag[16];
    #pragma unroll
    for (int c = 0; c < 16; ++c) {
        int d0 = c * 32;
        bf16x8 lo = *reinterpret_cast<const bf16x8*>(arow + d0 + half * 8);
        bf16x8 hi = *reinterpret_cast<const bf16x8*>(arow + d0 + 16 + half * 8);
        afrag[c] = combine8(lo, hi);
    }

    // online softmax state: 8 rows per lane (row = row_base + half*8 + v)
    float mrun[8], lrun[8], a0[8], a1[8], a2[8];
    #pragma unroll
    for (int v = 0; v < 8; ++v) { mrun[v] = -1e30f; lrun[v] = 0.f; a0[v] = a1[v] = a2[v] = 0.f; }

    const uint4* ksrcbase = reinterpret_cast<const uint4*>(KT + (size_t)b * NPTS * DKDIM);

#if USE_ASYNC_LDS
    // ---- CDNA5 async global->LDS double-buffer pipeline ----
    // Each thread issues 4 b128 async loads per 16 KB tile (ASYNCcnt +4/wave).
    {
        const uint4* g = ksrcbase + tid;
        uint4* l = reinterpret_cast<uint4*>(ktile[0]) + tid;
        #pragma unroll
        for (int i = 0; i < 4; ++i)
            __builtin_amdgcn_global_load_async_to_lds_b128(
                (__attribute__((address_space(1))) v4i*)(g + i * 256),
                (__attribute__((address_space(3))) v4i*)(l + i * 256), 0, 0);
    }
#else
    uint4 stage[4];
    {
        const uint4* g = ksrcbase + tid;
        #pragma unroll
        for (int i = 0; i < 4; ++i) stage[i] = g[i * 256];
    }
#endif

    for (int mt = 0; mt < NPTS / 16; ++mt) {
        const int buf = mt & 1;
#if USE_ASYNC_LDS
        if (mt < NPTS / 16 - 1) {
            // prefetch next tile into the other buffer (its readers finished
            // at the barrier that ended iteration mt-1)
            const uint4* g = ksrcbase + (size_t)(mt + 1) * 1024 + tid;
            uint4* l = reinterpret_cast<uint4*>(ktile[buf ^ 1]) + tid;
            #pragma unroll
            for (int i = 0; i < 4; ++i)
                __builtin_amdgcn_global_load_async_to_lds_b128(
                    (__attribute__((address_space(1))) v4i*)(g + i * 256),
                    (__attribute__((address_space(3))) v4i*)(l + i * 256), 0, 0);
            __builtin_amdgcn_s_wait_asynccnt(4);   // tile mt's 4 ops complete (in-order)
        } else {
            __builtin_amdgcn_s_wait_asynccnt(0);
        }
        __syncthreads();
#else
        {   // store the staged tile, then start fetching the next one
            uint4* l = reinterpret_cast<uint4*>(ktile[buf]) + tid;
            #pragma unroll
            for (int i = 0; i < 4; ++i) l[i * 256] = stage[i];
        }
        if (mt < NPTS / 16 - 1) {
            const uint4* g = ksrcbase + (size_t)(mt + 1) * 1024 + tid;
            #pragma unroll
            for (int i = 0; i < 4; ++i) stage[i] = g[i * 256];
        }
        __syncthreads();
#endif

        // S tile (16x16 f32) = Qtile . Ktile^T  via 16 bf16 WMMAs over DK
        f32x8 sacc = {0.f, 0.f, 0.f, 0.f, 0.f, 0.f, 0.f, 0.f};
        const unsigned short* krow = ktile[buf] + lr * DKDIM;
        #pragma unroll
        for (int c = 0; c < 16; ++c) {
            int o = c * 32 + half * 8;
            bf16x8 lo = *reinterpret_cast<const bf16x8*>(krow + o);
            bf16x8 hi = *reinterpret_cast<const bf16x8*>(krow + o + 16);
            bf16x16 bfrag = combine8(lo, hi);
            sacc = __builtin_amdgcn_wmma_f32_16x16x32_bf16(
                false, afrag[c], false, bfrag, (short)0, sacc, false, false);
        }

        // online softmax update + weighted accumulation of tgt (V, d_v = 3)
        const int col = mt * 16 + lr;
        const float t0 = tgts[col], t1 = tgts[NPTS + col], t2 = tgts[2 * NPTS + col];
        #pragma unroll
        for (int v = 0; v < 8; ++v) {
            float z    = scale * sacc[v];
            float tmax = redmax16(z);
            float mn   = fmaxf(mrun[v], tmax);
            float cf   = __expf(mrun[v] - mn);
            float p    = __expf(z - mn);
            mrun[v] = mn;
            lrun[v] = lrun[v] * cf + p;
            a0[v]   = a0[v]   * cf + p * t0;
            a1[v]   = a1[v]   * cf + p * t1;
            a2[v]   = a2[v]   * cf + p * t2;
        }
        __syncthreads();   // compute done before buf is overwritten (iter mt+1 writes buf)
    }

    // fold partial sums across the 16 lanes of each half-wave
    #pragma unroll
    for (int v = 0; v < 8; ++v) {
        lrun[v] = redsum16(lrun[v]);
        a0[v]   = redsum16(a0[v]);
        a1[v]   = redsum16(a1[v]);
        a2[v]   = redsum16(a2[v]);
    }

    // per-batch reductions: sum(src), sum(corr), sum(src_j * corr_k)
    if (lr == 0) {
        float ss[3] = {0, 0, 0}, sc[3] = {0, 0, 0}, cr[9] = {0, 0, 0, 0, 0, 0, 0, 0, 0};
        const float* srcb = src + (size_t)b * 3 * NPTS;
        #pragma unroll
        for (int v = 0; v < 8; ++v) {
            int row = row_base + half * 8 + v;
            float inv = 1.0f / lrun[v];
            float o0 = a0[v] * inv, o1 = a1[v] * inv, o2 = a2[v] * inv;
            float s0 = srcb[row], s1 = srcb[NPTS + row], s2 = srcb[2 * NPTS + row];
            ss[0] += s0; ss[1] += s1; ss[2] += s2;
            sc[0] += o0; sc[1] += o1; sc[2] += o2;
            cr[0] += s0 * o0; cr[1] += s0 * o1; cr[2] += s0 * o2;
            cr[3] += s1 * o0; cr[4] += s1 * o1; cr[5] += s1 * o2;
            cr[6] += s2 * o0; cr[7] += s2 * o1; cr[8] += s2 * o2;
        }
        float* rb = red + b * 16;
        #pragma unroll
        for (int j = 0; j < 3; ++j) { atomicAdd(rb + j, ss[j]); atomicAdd(rb + 3 + j, sc[j]); }
        #pragma unroll
        for (int j = 0; j < 9; ++j) atomicAdd(rb + 6 + j, cr[j]);
    }
}

// ---------------------------------------------------------------------------
// Kernel 3: per-batch 3x3 cross-covariance -> Jacobi SVD -> Kabsch R, t.
// 1 block, 16 threads (one per batch).
// ---------------------------------------------------------------------------
__global__ void svdhead_finalize_kernel(const float* __restrict__ red, float* __restrict__ out)
{
    const int b = threadIdx.x;
    if (b >= BATCH) return;
    const float* rb = red + b * 16;
    const float invN = 1.0f / (float)NPTS;

    float ms[3], mc[3];
    #pragma unroll
    for (int j = 0; j < 3; ++j) { ms[j] = rb[j] * invN; mc[j] = rb[3 + j] * invN; }

    float H[3][3];
    #pragma unroll
    for (int j = 0; j < 3; ++j)
        #pragma unroll
        for (int k = 0; k < 3; ++k)
            H[j][k] = rb[6 + 3 * j + k] - (float)NPTS * ms[j] * mc[k];

    // A = H^T H (symmetric); Jacobi eigen-decomposition -> V, eigenvalues
    float A[3][3], V[3][3] = {{1,0,0},{0,1,0},{0,0,1}};
    #pragma unroll
    for (int j = 0; j < 3; ++j)
        #pragma unroll
        for (int k = 0; k < 3; ++k)
            A[j][k] = H[0][j]*H[0][k] + H[1][j]*H[1][k] + H[2][j]*H[2][k];

    for (int sweep = 0; sweep < 12; ++sweep) {
        for (int pi = 0; pi < 3; ++pi) {
            const int p = (pi == 2) ? 1 : 0;
            const int q = (pi == 0) ? 1 : 2;
            float apq = A[p][q];
            if (fabsf(apq) < 1e-30f) continue;
            float tau = (A[q][q] - A[p][p]) / (2.0f * apq);
            float t = ((tau >= 0.0f) ? 1.0f : -1.0f) / (fabsf(tau) + sqrtf(1.0f + tau * tau));
            float c = 1.0f / sqrtf(1.0f + t * t);
            float s = t * c;
            for (int k = 0; k < 3; ++k) {
                float akp = A[k][p], akq = A[k][q];
                A[k][p] = c * akp - s * akq;
                A[k][q] = s * akp + c * akq;
            }
            for (int k = 0; k < 3; ++k) {
                float apk = A[p][k], aqk = A[q][k];
                A[p][k] = c * apk - s * aqk;
                A[q][k] = s * apk + c * aqk;
            }
            for (int k = 0; k < 3; ++k) {
                float vkp = V[k][p], vkq = V[k][q];
                V[k][p] = c * vkp - s * vkq;
                V[k][q] = s * vkp + c * vkq;
            }
        }
    }

    float sv[3]; int idx[3] = {0, 1, 2};
    #pragma unroll
    for (int i = 0; i < 3; ++i) sv[i] = sqrtf(fmaxf(A[i][i], 0.0f));
    if (sv[idx[0]] < sv[idx[1]]) { int t = idx[0]; idx[0] = idx[1]; idx[1] = t; }
    if (sv[idx[0]] < sv[idx[2]]) { int t = idx[0]; idx[0] = idx[2]; idx[2] = t; }
    if (sv[idx[1]] < sv[idx[2]]) { int t = idx[1]; idx[1] = idx[2]; idx[2] = t; }

    float Vs[3][3];
    #pragma unroll
    for (int j = 0; j < 3; ++j)
        #pragma unroll
        for (int i = 0; i < 3; ++i) Vs[j][i] = V[j][idx[i]];

    // Hv[:,i] = H @ Vs[:,i]
    float Hv[3][3];
    #pragma unroll
    for (int i = 0; i < 3; ++i)
        #pragma unroll
        for (int j = 0; j < 3; ++j)
            Hv[j][i] = H[j][0]*Vs[0][i] + H[j][1]*Vs[1][i] + H[j][2]*Vs[2][i];

    // U columns via normalize / Gram-Schmidt / cross-product handedness fix
    float U[3][3];
    float n0 = sqrtf(Hv[0][0]*Hv[0][0] + Hv[1][0]*Hv[1][0] + Hv[2][0]*Hv[2][0]);
    if (n0 > 1e-20f) { U[0][0]=Hv[0][0]/n0; U[1][0]=Hv[1][0]/n0; U[2][0]=Hv[2][0]/n0; }
    else             { U[0][0]=1.f; U[1][0]=0.f; U[2][0]=0.f; }
    float d01 = Hv[0][1]*U[0][0] + Hv[1][1]*U[1][0] + Hv[2][1]*U[2][0];
    float w0 = Hv[0][1]-d01*U[0][0], w1 = Hv[1][1]-d01*U[1][0], w2 = Hv[2][1]-d01*U[2][0];
    float n1 = sqrtf(w0*w0 + w1*w1 + w2*w2);
    if (n1 > 1e-20f) { U[0][1]=w0/n1; U[1][1]=w1/n1; U[2][1]=w2/n1; }
    else {
        float ex = (fabsf(U[0][0]) < 0.9f) ? 1.f : 0.f;
        float ey = 1.f - ex;
        float cx = U[1][0]*0.f - U[2][0]*ey;
        float cy = U[2][0]*ex - U[0][0]*0.f;
        float cz = U[0][0]*ey - U[1][0]*ex;
        float nn = sqrtf(cx*cx + cy*cy + cz*cz);
        U[0][1]=cx/nn; U[1][1]=cy/nn; U[2][1]=cz/nn;
    }
    float cpx = U[1][0]*U[2][1] - U[2][0]*U[1][1];
    float cpy = U[2][0]*U[0][1] - U[0][0]*U[2][1];
    float cpz = U[0][0]*U[1][1] - U[1][0]*U[0][1];
    float d2  = Hv[0][2]*cpx + Hv[1][2]*cpy + Hv[2][2]*cpz;
    float sgn = (d2 >= 0.0f) ? 1.0f : -1.0f;
    U[0][2] = sgn*cpx; U[1][2] = sgn*cpy; U[2][2] = sgn*cpz;

    float detU = sgn;   // det([u0 u1 u0xu1]) = 1, sign flip from u2
    float detV = Vs[0][0]*(Vs[1][1]*Vs[2][2]-Vs[1][2]*Vs[2][1])
               - Vs[0][1]*(Vs[1][0]*Vs[2][2]-Vs[1][2]*Vs[2][0])
               + Vs[0][2]*(Vs[1][0]*Vs[2][1]-Vs[1][1]*Vs[2][0]);
    float dd = (detU * detV >= 0.0f) ? 1.0f : -1.0f;

    float R[3][3];
    #pragma unroll
    for (int r = 0; r < 3; ++r)
        #pragma unroll
        for (int c2 = 0; c2 < 3; ++c2)
            R[r][c2] = Vs[r][0]*U[c2][0] + Vs[r][1]*U[c2][1] + dd*Vs[r][2]*U[c2][2];

    #pragma unroll
    for (int r = 0; r < 3; ++r)
        #pragma unroll
        for (int c2 = 0; c2 < 3; ++c2)
            out[b * 9 + 3 * r + c2] = R[r][c2];

    #pragma unroll
    for (int j = 0; j < 3; ++j) {
        float tj = mc[j] - (R[j][0]*ms[0] + R[j][1]*ms[1] + R[j][2]*ms[2]);
        out[BATCH * 9 + b * 3 + j] = tj;
    }
}

// ---------------------------------------------------------------------------
extern "C" void kernel_launch(void* const* d_in, const int* in_sizes, int n_in,
                              void* d_out, int out_size, void* d_ws, size_t ws_size,
                              hipStream_t stream) {
    (void)in_sizes; (void)n_in; (void)out_size; (void)ws_size;
    const float* src_emb = (const float*)d_in[0];
    const float* tgt_emb = (const float*)d_in[1];
    const float* src     = (const float*)d_in[2];
    const float* tgt     = (const float*)d_in[3];
    const float* temp    = (const float*)d_in[4];
    float* out = (float*)d_out;

    const size_t embElems = (size_t)BATCH * NPTS * DKDIM;
    unsigned short* QT = (unsigned short*)d_ws;
    unsigned short* KT = QT + embElems;
    float* red = (float*)(KT + embElems);

    zero_red_kernel<<<1, 256, 0, stream>>>(red);
    transpose_bf16_kernel<<<dim3(NPTS / 32, DKDIM / 32, 2 * BATCH), dim3(32, 8), 0, stream>>>(
        src_emb, tgt_emb, QT, KT);
    svdhead_main_kernel<<<dim3(NPTS / 128, BATCH), 256, 0, stream>>>(
        QT, KT, src, tgt, temp, red);
    svdhead_finalize_kernel<<<1, 16, 0, stream>>>(red, out);
}
